// SNNLayer_75368086110899
// MI455X (gfx1250) — matrix operand order repaired and checked by reading, not compiled, not run on hardware
//
#include <hip/hip_runtime.h>

#define IN_K   512
#define HIDDEN 512
#define OUT_N  256
#define TSTEPS 16
#define THRESH 0.5f
#define W2PAD  8            // 16B row pad -> rotates LDS banks across rows
#define MTILE  32           // rows per workgroup

typedef __attribute__((ext_vector_type(16))) _Float16 v16h;
typedef __attribute__((ext_vector_type(8)))  _Float16 v8h;
typedef __attribute__((ext_vector_type(8)))  float    v8f;

__device__ __forceinline__ v16h cat16(v8h lo, v8h hi) {
    return __builtin_shufflevector(lo, hi, 0,1,2,3,4,5,6,7,8,9,10,11,12,13,14,15);
}

// Prologue: convert W1 [512x512] and W2 [256x512] f32 -> f16 into workspace.
__global__ void snn_cvt_weights(const float* __restrict__ W1,
                                const float* __restrict__ W2,
                                _Float16* __restrict__ W1h,
                                _Float16* __restrict__ W2h) {
    int i = blockIdx.x * blockDim.x + threadIdx.x;
    const int n1 = HIDDEN * IN_K;     // 262144
    const int n2 = OUT_N * HIDDEN;    // 131072
    if (i < n1) {
        W1h[i] = (_Float16)W1[i];
    } else if (i < n1 + n2) {
        W2h[i - n1] = (_Float16)W2[i - n1];
    }
}

// Fused SNN. One workgroup = 8 waves = 32 rows. W2 lives in LDS for all 16
// timesteps (async-copied in once); spikes pass through a unioned LDS buffer;
// pre1/mem1/mem2/spike_sum never touch HBM.
__launch_bounds__(256, 1)
__global__ void snn_fused_kernel(const float* __restrict__ x,
                                 const float* __restrict__ b1,
                                 const float* __restrict__ b2,
                                 const _Float16* __restrict__ W1h,
                                 const _Float16* __restrict__ W2h,
                                 float* __restrict__ out) {
    // 32 KB: x tile (GEMM1 A) unioned with spike buffer (GEMM2 A)
    __shared__ _Float16 xsp[MTILE][IN_K];
    // 260 KB: whole W2 as f16, row-padded
    __shared__ _Float16 w2s[OUT_N][HIDDEN + W2PAD];

    const int tid   = threadIdx.x;
    const int lane  = tid & 31;
    const int wave  = tid >> 5;          // 0..7
    const int wcol  = wave & 3;          // column-split id
    const int mbase = (wave >> 2) * 16;  // row-group base (0 or 16)
    const int half_ = lane >> 4;         // 0/1
    const int ln16  = lane & 15;
    const long row0 = (long)blockIdx.x * MTILE;

    // ---- Async-stage all of W2 (f16) into LDS: 16384 x 16B chunks ----
    for (int idx = tid; idx < OUT_N * (HIDDEN / 8); idx += 256) {
        const int r = idx >> 6;            // row 0..255
        const int c = (idx & 63) << 3;     // col 0..504 step 8
        const uint64_t ga = (uint64_t)(uintptr_t)(W2h + r * HIDDEN + c);
        const uint32_t la = (uint32_t)(uintptr_t)&w2s[r][c];  // flat->LDS offset
        asm volatile("global_load_async_to_lds_b128 %0, %1, off"
                     :: "v"(la), "v"(ga) : "memory");
    }

    // ---- Stage x tile (32 rows x 512 f32 -> f16), overlapped with async W2 ----
    {
        const float4* xg = (const float4*)(x + row0 * IN_K);
        for (int idx = tid; idx < MTILE * (IN_K / 4); idx += 256) {
            float4 v = xg[idx];
            int r = idx / (IN_K / 4);
            int c = (idx % (IN_K / 4)) * 4;
            xsp[r][c + 0] = (_Float16)v.x;
            xsp[r][c + 1] = (_Float16)v.y;
            xsp[r][c + 2] = (_Float16)v.z;
            xsp[r][c + 3] = (_Float16)v.w;
        }
    }
    asm volatile("s_wait_asynccnt 0x0" ::: "memory");
    __syncthreads();

    // ---- GEMM1: pre1[16 x 128-per-wave] = x . W1^T + b1 (f32 accum) ----
    const int hbase = wcol * 128;
    v8f pre1[8];
    #pragma unroll
    for (int nt = 0; nt < 8; ++nt) {
        float bias = b1[hbase + nt * 16 + ln16];     // column-only
        #pragma unroll
        for (int e = 0; e < 8; ++e) pre1[nt][e] = bias;
    }

    for (int k0 = 0; k0 < IN_K; k0 += 32) {
        v8h alo = *(const v8h*)&xsp[mbase + ln16][k0 + half_ * 8];
        v8h ahi = *(const v8h*)&xsp[mbase + ln16][k0 + 16 + half_ * 8];
        v16h a  = cat16(alo, ahi);
        const int klo = k0 + half_ * 8;
        #pragma unroll
        for (int nt = 0; nt < 8; ++nt) {
            const _Float16* bp = W1h + (long)(hbase + nt * 16 + ln16) * IN_K + klo;
            v16h b = cat16(*(const v8h*)bp, *(const v8h*)(bp + 16));
            pre1[nt] = __builtin_amdgcn_wmma_f32_16x16x32_f16(
                false, a, false, b, (short)0, pre1[nt], false, false);
        }
    }
    __syncthreads();   // x tile no longer needed; xsp becomes the spike buffer

    // ---- Persistent LIF state in registers ----
    const int obase = wcol * 64;
    v8f mem1[8], mem2[4], ssum[4];
    #pragma unroll
    for (int nt = 0; nt < 8; ++nt)
        #pragma unroll
        for (int e = 0; e < 8; ++e) mem1[nt][e] = 0.0f;

    float bias2[4];
    #pragma unroll
    for (int ot = 0; ot < 4; ++ot) {
        bias2[ot] = b2[obase + ot * 16 + ln16];
        #pragma unroll
        for (int e = 0; e < 8; ++e) { mem2[ot][e] = 0.0f; ssum[ot][e] = 0.0f; }
    }

    // ---- Timestep recurrence ----
    for (int t = 0; t < TSTEPS; ++t) {
        // Layer-1 LIF: mem1 += pre1; spike; reset. Spikes -> LDS (f16, exact).
        #pragma unroll
        for (int nt = 0; nt < 8; ++nt) {
            const int col = hbase + nt * 16 + ln16;
            #pragma unroll
            for (int e = 0; e < 8; ++e) {
                float m = mem1[nt][e] + pre1[nt][e];
                bool s  = m > THRESH;
                mem1[nt][e] = s ? 0.0f : m;
                xsp[mbase + e + half_ * 8][col] = s ? (_Float16)1.0f : (_Float16)0.0f;
            }
        }
        __syncthreads();

        // Layer-2 GEMM: mem2 += spike1 . W2^T + b2  (A and B both from LDS)
        #pragma unroll
        for (int ot = 0; ot < 4; ++ot)
            #pragma unroll
            for (int e = 0; e < 8; ++e) mem2[ot][e] += bias2[ot];

        for (int k0 = 0; k0 < HIDDEN; k0 += 32) {
            v8h alo = *(const v8h*)&xsp[mbase + ln16][k0 + half_ * 8];
            v8h ahi = *(const v8h*)&xsp[mbase + ln16][k0 + 16 + half_ * 8];
            v16h a  = cat16(alo, ahi);
            const int klo = k0 + half_ * 8;
            #pragma unroll
            for (int ot = 0; ot < 4; ++ot) {
                const _Float16* wp = &w2s[obase + ot * 16 + ln16][klo];
                v16h b = cat16(*(const v8h*)wp, *(const v8h*)(wp + 16));
                mem2[ot] = __builtin_amdgcn_wmma_f32_16x16x32_f16(
                    false, a, false, b, (short)0, mem2[ot], false, false);
            }
        }

        // Layer-2 LIF: spike; accumulate; reset.
        #pragma unroll
        for (int ot = 0; ot < 4; ++ot)
            #pragma unroll
            for (int e = 0; e < 8; ++e) {
                float m = mem2[ot][e];
                bool s  = m > THRESH;
                ssum[ot][e] += s ? 1.0f : 0.0f;
                mem2[ot][e]  = s ? 0.0f : m;
            }
        __syncthreads();   // protect spike buffer before next timestep's writes
    }

    // ---- Epilogue: out = spike_sum / TIMESTEPS (C-layout scatter) ----
    #pragma unroll
    for (int ot = 0; ot < 4; ++ot) {
        const int col = obase + ot * 16 + ln16;
        #pragma unroll
        for (int e = 0; e < 8; ++e) {
            const long row = row0 + mbase + e + half_ * 8;
            out[row * OUT_N + col] = ssum[ot][e] * (1.0f / TSTEPS);
        }
    }
}

extern "C" void kernel_launch(void* const* d_in, const int* in_sizes, int n_in,
                              void* d_out, int out_size, void* d_ws, size_t ws_size,
                              hipStream_t stream) {
    const float* x  = (const float*)d_in[0];
    const float* W1 = (const float*)d_in[1];
    const float* b1 = (const float*)d_in[2];
    const float* W2 = (const float*)d_in[3];
    const float* b2 = (const float*)d_in[4];
    float* out = (float*)d_out;

    _Float16* W1h = (_Float16*)d_ws;
    _Float16* W2h = W1h + (size_t)HIDDEN * IN_K;

    const int ncvt = HIDDEN * IN_K + OUT_N * HIDDEN;   // 393216
    snn_cvt_weights<<<(ncvt + 255) / 256, 256, 0, stream>>>(W1, W2, W1h, W2h);

    const int rows = in_sizes[0] / IN_K;               // B*S = 32768
    snn_fused_kernel<<<rows / MTILE, 256, 0, stream>>>(x, b1, b2, W1h, W2h, out);
}
